// Dgm_61770219651304
// MI455X (gfx1250) — matrix operand order repaired and checked by reading, not compiled, not run on hardware
//
#include <hip/hip_runtime.h>
#include <stdint.h>

#define B_ 4096
#define D_ 1024
#define H_ 2048
#define L_ 256
#define T_ 8

typedef __attribute__((ext_vector_type(16))) __bf16 v16bf;
typedef __attribute__((ext_vector_type(8)))  __bf16 v8bf;
typedef __attribute__((ext_vector_type(8)))  float  v8f;
typedef __attribute__((ext_vector_type(4)))  unsigned int u32x4;
typedef __attribute__((ext_vector_type(4)))  int i32x4;
typedef __attribute__((ext_vector_type(8)))  int i32x8;

#if defined(__has_builtin)
#  if __has_builtin(__builtin_amdgcn_tensor_load_to_lds) && __has_builtin(__builtin_amdgcn_s_wait_tensorcnt)
#    define HAS_TDM 1
#  endif
#endif
#ifndef HAS_TDM
#  define HAS_TDM 0
#endif

// LDS tile geometry: 128 rows x 64 bf16 data (+8 pad) -> 72 elems / 144 B per row
#define TROW 72

#if HAS_TDM
// One 2D TDM tile: 128 rows x 64 bf16, global row stride = strideK elems,
// rows >= validRows read as zero (OOB). LDS padded +16B per 128B row.
__device__ __forceinline__ void tdm_load_tile(const __bf16* gsrc, unsigned ldsoff,
                                              int strideK, int validRows) {
  uint64_t ga = (uint64_t)(uintptr_t)gsrc;
  u32x4 g0;
  g0[0] = 1u;                                                  // count=1, user mode
  g0[1] = ldsoff;                                              // lds_addr
  g0[2] = (unsigned)(ga & 0xFFFFFFFFu);                        // global_addr lo
  g0[3] = (unsigned)((ga >> 32) & 0x01FFFFFFu) | 0x80000000u;  // hi | type=2
  i32x8 g1;
  // data_size=2B, pad_enable, pad_interval=32DW(code4), pad_amount=4DW(code3)
  g1[0] = (1 << 16) | (1 << 20) | (4 << 22) | (3 << 25);
  g1[1] = (int)(((unsigned)strideK & 0xFFFFu) << 16);                        // tensor_dim0 lo16
  g1[2] = (int)(((unsigned)strideK >> 16) | (((unsigned)validRows) << 16));  // dim0 hi | dim1 lo16
  g1[3] = (int)((((unsigned)validRows >> 16) & 0xFFFFu) | (64u << 16));      // dim1 hi | tile_dim0=64
  g1[4] = 128;                                                 // tile_dim1 = 128
  g1[5] = strideK;                                             // tensor_dim0_stride
  g1[6] = 0;
  g1[7] = 0;
  i32x4 gz = {0, 0, 0, 0};
#if __clang_major__ >= 23
  i32x8 gz8 = {0, 0, 0, 0, 0, 0, 0, 0};
  __builtin_amdgcn_tensor_load_to_lds(g0, g1, gz, gz, gz8, 0);
#else
  __builtin_amdgcn_tensor_load_to_lds(g0, g1, gz, gz, 0);
#endif
}
#endif

// ---------------- permutation building (group rows by task) ----------------

__global__ void k_init(int* counts, int* cursor) {
  if (threadIdx.x < T_) { counts[threadIdx.x] = 0; cursor[threadIdx.x] = 0; }
}

__global__ void k_count(const int* __restrict__ task, int* __restrict__ counts) {
  int i = blockIdx.x * blockDim.x + threadIdx.x;
  if (i < B_) atomicAdd(&counts[task[i]], 1);
}

__global__ void k_scan(const int* __restrict__ counts, int* __restrict__ offs,
                       int* __restrict__ cursor) {
  if (threadIdx.x == 0 && blockIdx.x == 0) {
    int run = 0;
    for (int t = 0; t < T_; ++t) { offs[t] = run; cursor[t] = run; run += counts[t]; }
  }
}

__global__ void k_scatter(const int* __restrict__ task, int* __restrict__ cursor,
                          int* __restrict__ perm) {
  int i = blockIdx.x * blockDim.x + threadIdx.x;
  if (i < B_) { int p = atomicAdd(&cursor[task[i]], 1); perm[p] = i; }
}

// ---------------- x: fp32 -> bf16 row-major copy ----------------

__global__ void k_xconv(const float* __restrict__ x, __bf16* __restrict__ xbf) {
  int i = blockIdx.x * blockDim.x + threadIdx.x;      // over B_*D_/4
  float4 v = *(const float4*)(x + (size_t)i * 4);
  __align__(8) __bf16 b[4] = {(__bf16)v.x, (__bf16)v.y, (__bf16)v.z, (__bf16)v.w};
  *(uint2*)(xbf + (size_t)i * 4) = *(const uint2*)b;
}

// ---------------- weight pass: (tasks,K,N) fp32 -> (tasks,N,K) bf16 ----------------
// grid: x = N/64, y = K/64, z = task

__global__ __launch_bounds__(256) void k_wtrans(const float* __restrict__ W,
                                                __bf16* __restrict__ Wt, int K, int N) {
  __shared__ float tile[64][65];
  const float* Wz = W  + (size_t)blockIdx.z * K * N;
  __bf16*      Wo = Wt + (size_t)blockIdx.z * K * N;
  const int k0 = blockIdx.y * 64, n0 = blockIdx.x * 64;
  const int tid = threadIdx.x;
  const int rr = tid >> 4, cc = (tid & 15) * 4;
#pragma unroll
  for (int it = 0; it < 4; ++it) {
    int r = rr + it * 16;
    float4 v = *(const float4*)(Wz + (size_t)(k0 + r) * N + n0 + cc);
    tile[r][cc + 0] = v.x; tile[r][cc + 1] = v.y;
    tile[r][cc + 2] = v.z; tile[r][cc + 3] = v.w;
  }
  __syncthreads();
  const int n = tid >> 2, koff = (tid & 3) * 16;
  __align__(16) __bf16 buf[16];
#pragma unroll
  for (int j = 0; j < 16; ++j) buf[j] = (__bf16)tile[koff + j][n];
  __bf16* dst = Wo + (size_t)(n0 + n) * K + k0 + koff;
  ((uint4*)dst)[0] = ((const uint4*)buf)[0];
  ((uint4*)dst)[1] = ((const uint4*)buf)[1];
}

// ---------------- reparameterization (bf16 in/out) ----------------

__global__ void k_reparam(const __bf16* __restrict__ scat, const float* __restrict__ eps,
                          const int* __restrict__ perm, __bf16* __restrict__ z) {
  int i = blockIdx.x * blockDim.x + threadIdx.x;      // over B_*L_
  int r = i >> 8, l = i & (L_ - 1);
  int src = perm[r];
  float mu = (float)scat[(size_t)r * (2 * L_) + l];
  float ls = (float)scat[(size_t)r * (2 * L_) + L_ + l];
  z[i] = (__bf16)(mu + __expf(ls) * eps[(size_t)src * L_ + l]);
}

// ---------------- grouped WMMA GEMM ----------------
// A: bf16 (rows x K) row-major. Wt: bf16 (N x K) per task (pre-transposed).
// BM=128, BN=128, BK=64; 8 waves, each 32(m) x 64(n) = 2x4 wmma tiles, 2 K-substeps.
// Double-buffered LDS; A and B tiles staged by Tensor Data Mover where possible
// (OOB rows auto-zero via tensor_dim1 = cnt - row0); gather layer stages A manually.

template<int ACT, bool GATHER_A, bool SCATTER_C, bool GROUPED, bool OUTBF>
__global__ __launch_bounds__(256)
void gemm_wmma(const __bf16* __restrict__ A, const __bf16* __restrict__ Wt,
               const float* __restrict__ bias, void* __restrict__ CoutV,
               const int* __restrict__ perm, const int* __restrict__ offs,
               const int* __restrict__ counts, int K, int N)
{
  const int t = GROUPED ? blockIdx.z : 0;
  const int groupStart = GROUPED ? offs[t]   : 0;
  const int cnt        = GROUPED ? counts[t] : B_;
  const int row0 = blockIdx.y * 128;
  if (row0 >= cnt) return;                 // uniform
  const int n0 = blockIdx.x * 128;

  const __bf16* Wtt   = Wt   + (size_t)t * K * N;
  const float*  biasT = bias + (size_t)t * N;

  __shared__ __bf16 Ash[2][128 * TROW];
  __shared__ __bf16 Bsh[2][128 * TROW];

  const int tid  = threadIdx.x;
  const int lane = tid & 31;
  const int l15  = lane & 15;
  const int half = lane >> 4;
  const int wave = tid >> 5;
  const int wm   = wave & 3;
  const int wn   = wave >> 2;

  constexpr bool TDM_A = (HAS_TDM != 0) && !GATHER_A;

  // ---- manual A staging state (gather layer / no-TDM): 4 x 16B chunks ----
  const __bf16* asrc[4]; int alds[4]; bool avalid[4];
  if constexpr (!TDM_A) {
#pragma unroll
    for (int it = 0; it < 4; ++it) {
      int c = tid + it * 256;
      int r = c >> 3, seg = c & 7;
      bool valid = (row0 + r) < cnt;
      int gr = groupStart + row0 + r;
      int srcRow = GATHER_A ? (valid ? perm[gr] : 0) : (valid ? gr : 0);
      asrc[it]   = A + (size_t)srcRow * K + seg * 8;
      alds[it]   = r * TROW + seg * 8;
      avalid[it] = valid;
    }
  }
#if !HAS_TDM
  const __bf16* bsrc[4]; int blds[4];
#pragma unroll
  for (int it = 0; it < 4; ++it) {
    int c = tid + it * 256;
    int r = c >> 3, seg = c & 7;
    bsrc[it] = Wtt + (size_t)(n0 + r) * K + seg * 8;
    blds[it] = r * TROW + seg * 8;
  }
#endif

  auto stage = [&](int buf, int k0) {
    if constexpr (TDM_A) {
#if HAS_TDM
      if (tid == 0)
        tdm_load_tile(A + (size_t)(groupStart + row0) * K + k0,
                      (unsigned)(uintptr_t)&Ash[buf][0], K, cnt - row0);
#endif
    } else {
#pragma unroll
      for (int it = 0; it < 4; ++it) {
        uint4 v = make_uint4(0u, 0u, 0u, 0u);
        if (avalid[it]) v = *(const uint4*)(asrc[it] + k0);
        *(uint4*)&Ash[buf][alds[it]] = v;
      }
    }
#if HAS_TDM
    if (tid == 0)
      tdm_load_tile(Wtt + (size_t)n0 * K + k0,
                    (unsigned)(uintptr_t)&Bsh[buf][0], K, 128);
#else
#pragma unroll
    for (int it = 0; it < 4; ++it)
      *(uint4*)&Bsh[buf][blds[it]] = *(const uint4*)(bsrc[it] + k0);
#endif
  };

  v8f acc[2][4] = {};
  stage(0, 0);
  int cur = 0;

  for (int k0 = 0; k0 < K; k0 += 64) {
#if HAS_TDM
    if (tid == 0) __builtin_amdgcn_s_wait_tensorcnt(0);
#endif
    __syncthreads();
    if (k0 + 64 < K) stage(cur ^ 1, k0 + 64);

#pragma unroll
    for (int ks = 0; ks < 2; ++ks) {
      v16bf af[2], bfr[4];
#pragma unroll
      for (int mi = 0; mi < 2; ++mi) {
        const __bf16* p = &Ash[cur][(wm * 32 + mi * 16 + l15) * TROW + ks * 32 + half * 8];
        v8bf lo = *(const v8bf*)p;
        v8bf hi = *(const v8bf*)(p + 16);
        af[mi] = __builtin_shufflevector(lo, hi, 0,1,2,3,4,5,6,7,8,9,10,11,12,13,14,15);
      }
#pragma unroll
      for (int ni = 0; ni < 4; ++ni) {
        const __bf16* p = &Bsh[cur][(wn * 64 + ni * 16 + l15) * TROW + ks * 32 + half * 8];
        v8bf lo = *(const v8bf*)p;
        v8bf hi = *(const v8bf*)(p + 16);
        bfr[ni] = __builtin_shufflevector(lo, hi, 0,1,2,3,4,5,6,7,8,9,10,11,12,13,14,15);
      }
#pragma unroll
      for (int mi = 0; mi < 2; ++mi)
#pragma unroll
        for (int ni = 0; ni < 4; ++ni)
          acc[mi][ni] = __builtin_amdgcn_wmma_f32_16x16x32_bf16(
              false, af[mi], false, bfr[ni], (short)0, acc[mi][ni], false, false);
    }
    cur ^= 1;
  }

  // ---- epilogue ----
  float*  Cf = (float*)CoutV;
  __bf16* Cb = (__bf16*)CoutV;
  float bv[4];
#pragma unroll
  for (int ni = 0; ni < 4; ++ni) bv[ni] = biasT[n0 + wn * 64 + ni * 16 + l15];
#pragma unroll
  for (int mi = 0; mi < 2; ++mi) {
#pragma unroll
    for (int v = 0; v < 8; ++v) {
      int rl = wm * 32 + mi * 16 + half * 8 + v;
      if (row0 + rl < cnt) {
        int gr  = groupStart + row0 + rl;
        int dst = SCATTER_C ? perm[gr] : gr;
#pragma unroll
        for (int ni = 0; ni < 4; ++ni) {
          int col = n0 + wn * 64 + ni * 16 + l15;
          float val = acc[mi][ni][v] + bv[ni];
          if (ACT == 1) val = fmaxf(val, 0.f);
          else if (ACT == 2) val = 1.f / (1.f + __expf(-val));
          if (OUTBF) Cb[(size_t)dst * N + col] = (__bf16)val;
          else       Cf[(size_t)dst * N + col] = val;
        }
      }
    }
  }
}

// ---------------- driver ----------------

extern "C" void kernel_launch(void* const* d_in, const int* in_sizes, int n_in,
                              void* d_out, int out_size, void* d_ws, size_t ws_size,
                              hipStream_t stream) {
  const float* x      = (const float*)d_in[0];
  const int*   task   = (const int*)  d_in[1];
  const float* eps    = (const float*)d_in[2];
  const float* enc_W1 = (const float*)d_in[3];
  const float* enc_b1 = (const float*)d_in[4];
  const float* enc_W2 = (const float*)d_in[5];
  const float* enc_b2 = (const float*)d_in[6];
  const float* enc_W3 = (const float*)d_in[7];
  const float* enc_b3 = (const float*)d_in[8];
  const float* enc_W4 = (const float*)d_in[9];
  const float* enc_b4 = (const float*)d_in[10];
  const float* ds_W1  = (const float*)d_in[11];
  const float* ds_b1  = (const float*)d_in[12];
  const float* ds_W2  = (const float*)d_in[13];
  const float* ds_b2  = (const float*)d_in[14];
  const float* hd_W1  = (const float*)d_in[15];
  const float* hd_b1  = (const float*)d_in[16];
  const float* hd_W2  = (const float*)d_in[17];
  const float* hd_b2  = (const float*)d_in[18];

  char* ws = (char*)d_ws;
  int* counts = (int*)(ws + 0);
  int* offs   = (int*)(ws + 64);
  int* cursor = (int*)(ws + 128);
  int* perm   = (int*)(ws + 256);
  size_t off = 32768;
  __bf16* xbf   = (__bf16*)(ws + off); off += (size_t)B_ * D_ * 2;
  __bf16* h1b   = (__bf16*)(ws + off); off += (size_t)B_ * H_ * 2;
  __bf16* h2b   = (__bf16*)(ws + off); off += (size_t)B_ * H_ * 2;
  __bf16* scatb = (__bf16*)(ws + off); off += (size_t)B_ * 2 * L_ * 2;
  __bf16* zb    = (__bf16*)(ws + off); off += (size_t)B_ * L_ * 2;
  __bf16* wbuf  = (__bf16*)(ws + off);   // 64 MB slab, reused per layer

  dim3 blk(256);
  // permutation
  k_init   <<<1, 32, 0, stream>>>(counts, cursor);
  k_count  <<<B_ / 256, 256, 0, stream>>>(task, counts);
  k_scan   <<<1, 32, 0, stream>>>(counts, offs, cursor);
  k_scatter<<<B_ / 256, 256, 0, stream>>>(task, cursor, perm);
  // x -> bf16
  k_xconv  <<<(B_ * D_ / 4) / 256, 256, 0, stream>>>(x, xbf);

  // encoder
  k_wtrans<<<dim3(H_ / 64, D_ / 64, T_), blk, 0, stream>>>(enc_W1, wbuf, D_, H_);
  gemm_wmma<1, true,  false, true,  true ><<<dim3(H_ / 128, B_ / 128, T_), blk, 0, stream>>>(
      xbf, wbuf, enc_b1, h1b, perm, offs, counts, D_, H_);
  k_wtrans<<<dim3(H_ / 64, H_ / 64, T_), blk, 0, stream>>>(enc_W2, wbuf, H_, H_);
  gemm_wmma<1, false, false, true,  true ><<<dim3(H_ / 128, B_ / 128, T_), blk, 0, stream>>>(
      h1b, wbuf, enc_b2, h2b, perm, offs, counts, H_, H_);
  k_wtrans<<<dim3(H_ / 64, H_ / 64, T_), blk, 0, stream>>>(enc_W3, wbuf, H_, H_);
  gemm_wmma<1, false, false, true,  true ><<<dim3(H_ / 128, B_ / 128, T_), blk, 0, stream>>>(
      h2b, wbuf, enc_b3, h1b, perm, offs, counts, H_, H_);
  k_wtrans<<<dim3((2 * L_) / 64, H_ / 64, T_), blk, 0, stream>>>(enc_W4, wbuf, H_, 2 * L_);
  gemm_wmma<0, false, false, true,  true ><<<dim3((2 * L_) / 128, B_ / 128, T_), blk, 0, stream>>>(
      h1b, wbuf, enc_b4, scatb, perm, offs, counts, H_, 2 * L_);
  // z = mu + exp(log_sigma) * eps[perm]
  k_reparam<<<(B_ * L_) / 256, 256, 0, stream>>>(scatb, eps, perm, zb);
  // shared decoder stem
  k_wtrans<<<dim3(H_ / 64, L_ / 64, 1), blk, 0, stream>>>(ds_W1, wbuf, L_, H_);
  gemm_wmma<1, false, false, false, true ><<<dim3(H_ / 128, B_ / 128, 1), blk, 0, stream>>>(
      zb, wbuf, ds_b1, h2b, perm, offs, counts, L_, H_);
  k_wtrans<<<dim3(H_ / 64, H_ / 64, 1), blk, 0, stream>>>(ds_W2, wbuf, H_, H_);
  gemm_wmma<1, false, false, false, true ><<<dim3(H_ / 128, B_ / 128, 1), blk, 0, stream>>>(
      h2b, wbuf, ds_b2, h1b, perm, offs, counts, H_, H_);
  // per-task heads
  k_wtrans<<<dim3(H_ / 64, H_ / 64, T_), blk, 0, stream>>>(hd_W1, wbuf, H_, H_);
  gemm_wmma<1, false, false, true,  true ><<<dim3(H_ / 128, B_ / 128, T_), blk, 0, stream>>>(
      h1b, wbuf, hd_b1, h2b, perm, offs, counts, H_, H_);
  k_wtrans<<<dim3(D_ / 64, H_ / 64, T_), blk, 0, stream>>>(hd_W2, wbuf, H_, D_);
  gemm_wmma<2, false, true,  true,  false><<<dim3(D_ / 128, B_ / 128, T_), blk, 0, stream>>>(
      h2b, wbuf, hd_b2, d_out, perm, offs, counts, H_, D_);
}
